// HGNN2_16466904613536
// MI455X (gfx1250) — matrix operand order repaired and checked by reading, not compiled, not run on hardware
//
#include <hip/hip_runtime.h>
#include <hip/hip_bf16.h>
#include <math.h>

// ---------------------------------------------------------------------------
// HGNN forward for MI455X (gfx1250, wave32).
//   * Node GEMMs [N,32]@[32,32] via v_wmma_f32_16x16x32_f16 (one wave = 16 rows,
//     one K=32 WMMA per 16-col tile; f16 in / f32 accumulate).
//   * Edge softmax-attention: 2 passes (atomic max, atomic accumulate),
//     normalization folded into per-node divide.  Wave-per-edge gather/scatter.
//   * All launches on `stream`; scratch carved out of d_ws; no capture-illegal API.
// ---------------------------------------------------------------------------

#define NEG_SLOPE 0.2f
#define LN_EPS    1e-5f
#define HC        32      // hidden channels
#define NHEADS    5
#define NB        64      // graphs in batch

typedef __attribute__((ext_vector_type(16))) _Float16 v16h;
typedef __attribute__((ext_vector_type(8)))  float    v8f;

// ------------------------------- device helpers ----------------------------

__device__ __forceinline__ void atomicMaxF(float* addr, float val) {
  unsigned int* ua = (unsigned int*)addr;
  unsigned int old = *ua;
  while (__uint_as_float(old) < val) {
    unsigned int assumed = old;
    old = atomicCAS(ua, assumed, __float_as_uint(val));
    if (old == assumed) break;
  }
}

__device__ __forceinline__ float lrelu(float x) {
  return x > 0.0f ? x : NEG_SLOPE * x;
}
__device__ __forceinline__ float elu1(float x) {
  return x > 0.0f ? x : (__expf(x) - 1.0f);
}

// ------------------------------- kernels -----------------------------------

__global__ void fill_f32(float* __restrict__ p, float v, int n) {
  int i = blockIdx.x * blockDim.x + threadIdx.x;
  if (i < n) p[i] = v;
}

// Y[N,32] = X[N,32] @ W[32,32]   (W row-major: W[k*32+j], matching x @ W)
// One wave per 16-row tile.  A: 16x32 f16 layout (lane halves split K by 8/16).
// B: 32x16 f16, same K order as A: VGPR v holds K=2v..2v+1 (+8 for v>=4), lane-half adds +8 to K.
// C/D: v8f, M split across lane halves.
// Tile-uniform tail handling: full tiles take a branch-free load/store path.
__global__ __launch_bounds__(32)
void gemm32_wmma(const float* __restrict__ X, const float* __restrict__ W,
                 float* __restrict__ Y, int N) {
  const int lane = threadIdx.x;
  const int row0 = blockIdx.x * 16;
  const int m    = lane & 15;
  const int hi   = lane >> 4;
  const bool full = (row0 + 16 <= N);          // wave-uniform

  int row = row0 + m;
  if (!full && row >= N) row = N - 1;          // clamp only on tail tile

  v16h a;
  #pragma unroll
  for (int j = 0; j < 16; ++j) {
    int k = 16 * (j >> 3) + 8 * hi + (j & 7);
    a[j] = (_Float16)X[row * HC + k];
  }
  v16h b0, b1;
  #pragma unroll
  for (int j = 0; j < 16; ++j) {
    int k = 16 * (j >> 3) + 8 * hi + (j & 7);
    b0[j] = (_Float16)W[k * HC + m];        // output cols 0..15
    b1[j] = (_Float16)W[k * HC + 16 + m];   // output cols 16..31
  }
  v8f c0 = {}, c1 = {};
  c0 = __builtin_amdgcn_wmma_f32_16x16x32_f16(false, a, false, b0, (short)0, c0, false, false);
  c1 = __builtin_amdgcn_wmma_f32_16x16x32_f16(false, a, false, b1, (short)0, c1, false, false);

  const int mb = hi * 8;
  if (full) {
    #pragma unroll
    for (int r = 0; r < 8; ++r) {
      int orow = row0 + mb + r;
      Y[orow * HC + m]      = c0[r];
      Y[orow * HC + 16 + m] = c1[r];
    }
  } else {
    #pragma unroll
    for (int r = 0; r < 8; ++r) {
      int orow = row0 + mb + r;
      if (orow < N) {
        Y[orow * HC + m]      = c0[r];
        Y[orow * HC + 16 + m] = c1[r];
      }
    }
  }
}

// s[n] = dot(Y[n,:], a[:])
__global__ void dot_vec(const float* __restrict__ Y, const float* __restrict__ a,
                        float* __restrict__ s, int N) {
  int n = blockIdx.x * blockDim.x + threadIdx.x;
  if (n >= N) return;
  float acc = 0.0f;
  #pragma unroll
  for (int c = 0; c < HC; ++c) acc += Y[n * HC + c] * a[c];
  s[n] = acc;
}

// pass 1: per-edge e = lrelu(ss[src]+sd[dst]); segment max into m[dst]
__global__ void edge_max(const int* __restrict__ src, const int* __restrict__ dst,
                         const float* __restrict__ ss, const float* __restrict__ sd,
                         float* __restrict__ m, int E) {
  int i = blockIdx.x * blockDim.x + threadIdx.x;
  if (i >= E) return;
  float e = lrelu(ss[src[i]] + sd[dst[i]]);
  atomicMaxF(&m[dst[i]], e);
}

// m = finite(m) ? m : 0   (nodes with no in-edges)
__global__ void fix_m(float* __restrict__ m, int n) {
  int i = blockIdx.x * blockDim.x + threadIdx.x;
  if (i >= n) return;
  float v = m[i];
  if (!(v > -3.0e38f)) m[i] = 0.0f;   // catches -inf and NaN
}

// pass 2: wave per edge; lane = channel.  unnorm[dst] += hs[src]*w ; den[dst] += w
__global__ void edge_accum(const int* __restrict__ src, const int* __restrict__ dst,
                           const float* __restrict__ ss, const float* __restrict__ sd,
                           const float* __restrict__ m,  const float* __restrict__ hs,
                           float* __restrict__ unnorm, float* __restrict__ den, int E) {
  int wave = (int)((blockIdx.x * (unsigned)blockDim.x + threadIdx.x) >> 5);
  int lane = threadIdx.x & 31;
  if (wave >= E) return;
  int s = src[wave], d = dst[wave];
  __builtin_prefetch(&hs[s * HC], 0, 1);              // global_prefetch_b8
  float e = lrelu(ss[s] + sd[d]);
  float w = __expf(e - m[d]);
  atomicAdd(&unnorm[d * HC + lane], hs[s * HC + lane] * w);
  if (lane == 0) atomicAdd(&den[d], w);
}

// comb[n,c] += unnorm[n,c]/max(den[n],1e-16) + bias[c]
__global__ void gat_combine(float* __restrict__ comb, const float* __restrict__ unnorm,
                            const float* __restrict__ den, const float* __restrict__ bias,
                            int Nd) {
  int idx = blockIdx.x * blockDim.x + threadIdx.x;
  if (idx >= Nd * HC) return;
  int n = idx >> 5, c = idx & 31;
  float dn = den[n];
  dn = dn > 1e-16f ? dn : 1e-16f;
  comb[idx] += unnorm[idx] / dn + bias[c];
}

// x_out[n,:] = LayerNorm(elu(comb[n,:])) * g + bt ; wave per node, lane = channel
__global__ void elu_ln(float* __restrict__ xout, const float* __restrict__ comb,
                       const float* __restrict__ g, const float* __restrict__ bt, int N) {
  int wave = (int)((blockIdx.x * (unsigned)blockDim.x + threadIdx.x) >> 5);
  int c = threadIdx.x & 31;
  if (wave >= N) return;
  float x = elu1(comb[wave * HC + c]);
  float s = x;
  for (int o = 16; o > 0; o >>= 1) s += __shfl_xor(s, o, 32);
  float mu = s * (1.0f / HC);
  float d = x - mu;
  float v = d * d;
  for (int o = 16; o > 0; o >>= 1) v += __shfl_xor(v, o, 32);
  v *= (1.0f / HC);
  xout[wave * HC + c] = d * rsqrtf(v + LN_EPS) * g[c] + bt[c];
}

// attention logits l[n,h] = xf[n,:]@attW[:,h] + attb[h]; track per-head max
__global__ void att_logits(const float* __restrict__ xf, const float* __restrict__ W,
                           const float* __restrict__ b, float* __restrict__ l,
                           float* __restrict__ hmax, int Nf) {
  int n = blockIdx.x * blockDim.x + threadIdx.x;
  if (n >= Nf) return;
  float x[HC];
  #pragma unroll
  for (int c = 0; c < HC; ++c) x[c] = xf[n * HC + c];
  #pragma unroll
  for (int h = 0; h < NHEADS; ++h) {
    float acc = b[h];
    #pragma unroll
    for (int c = 0; c < HC; ++c) acc += x[c] * W[c * NHEADS + h];
    l[n * NHEADS + h] = acc;
    atomicMaxF(&hmax[h], acc);
  }
}

// l := exp(l - hmax[h]); hsum[h] += l
__global__ void att_exp(float* __restrict__ l, const float* __restrict__ hmax,
                        float* __restrict__ hsum, int total) {
  int i = blockIdx.x * blockDim.x + threadIdx.x;
  if (i >= total) return;
  int h = i % NHEADS;
  float w = __expf(l[i] - hmax[h]);
  l[i] = w;
  atomicAdd(&hsum[h], w);
}

// pooled[b, h, c] += (w[n,h]/hsum[h]) * xf[n,c] ; wave per node
__global__ void att_pool(const float* __restrict__ w, const float* __restrict__ hsum,
                         const float* __restrict__ xf, const int* __restrict__ batch,
                         float* __restrict__ pooled, int Nf) {
  int wave = (int)((blockIdx.x * (unsigned)blockDim.x + threadIdx.x) >> 5);
  int c = threadIdx.x & 31;
  if (wave >= Nf) return;
  int b = batch[wave];
  float xv = xf[wave * HC + c];
  #pragma unroll
  for (int h = 0; h < NHEADS; ++h) {
    float alpha = w[wave * NHEADS + h] / hsum[h];
    atomicAdd(&pooled[(b * NHEADS + h) * HC + c], alpha * xv);
  }
}

// generic small dense: Y[N,M] = X[N,K]@W[K,M] + b ; act 1 = elu
__global__ void dense(const float* __restrict__ X, const float* __restrict__ W,
                      const float* __restrict__ b, float* __restrict__ Y,
                      int N, int K, int M, int act) {
  int idx = blockIdx.x * blockDim.x + threadIdx.x;
  if (idx >= N * M) return;
  int n = idx / M, j = idx % M;
  float acc = b[j];
  for (int k = 0; k < K; ++k) acc += X[n * K + k] * W[k * M + j];
  if (act) acc = elu1(acc);
  Y[idx] = acc;
}

// out = concat(lcb[64], ucb[64], mu[64])
__global__ void final_out(const float* __restrict__ muv, const float* __restrict__ piv,
                          float* __restrict__ out) {
  int i = threadIdx.x;
  if (i >= NB) return;
  float m = muv[i];
  out[i]            = m - __expf(piv[i * 2 + 0]);
  out[NB + i]       = m + __expf(piv[i * 2 + 1]);
  out[2 * NB + i]   = m;
}

// ------------------------------- host side ---------------------------------

static void run_gat(hipStream_t stream,
                    const float* xsrc, int Ns, const float* xdst, int Nd,
                    const int* ei, int E,
                    const float* Ws, const float* Wd,
                    const float* as, const float* ad, const float* bias,
                    float* comb,
                    float* hs, float* hd, float* ss, float* sd,
                    float* m, float* den, float* unnorm) {
  const int* src = ei;
  const int* dst = ei + E;
  gemm32_wmma<<<(Ns + 15) / 16, 32, 0, stream>>>(xsrc, Ws, hs, Ns);
  gemm32_wmma<<<(Nd + 15) / 16, 32, 0, stream>>>(xdst, Wd, hd, Nd);
  dot_vec<<<(Ns + 255) / 256, 256, 0, stream>>>(hs, as, ss, Ns);
  dot_vec<<<(Nd + 255) / 256, 256, 0, stream>>>(hd, ad, sd, Nd);
  fill_f32<<<(Nd + 255) / 256, 256, 0, stream>>>(m, -__builtin_inff(), Nd);
  fill_f32<<<(Nd + 255) / 256, 256, 0, stream>>>(den, 0.0f, Nd);
  fill_f32<<<(Nd * HC + 255) / 256, 256, 0, stream>>>(unnorm, 0.0f, Nd * HC);
  edge_max<<<(E + 255) / 256, 256, 0, stream>>>(src, dst, ss, sd, m, E);
  fix_m<<<(Nd + 255) / 256, 256, 0, stream>>>(m, Nd);
  edge_accum<<<((size_t)E * 32 + 255) / 256, 256, 0, stream>>>(src, dst, ss, sd, m, hs, unnorm, den, E);
  gat_combine<<<(Nd * HC + 255) / 256, 256, 0, stream>>>(comb, unnorm, den, bias, Nd);
}

extern "C" void kernel_launch(void* const* d_in, const int* in_sizes, int n_in,
                              void* d_out, int out_size, void* d_ws, size_t ws_size,
                              hipStream_t stream) {
  // ----- input layout (setup_inputs() dict insertion order, nested dicts
  //        flattened in insertion order; lists in index order) -----
  const int Ni = in_sizes[0] / HC;
  const int Nf = in_sizes[1] / HC;
  const int No = in_sizes[2] / HC;
  const int E_if = in_sizes[3] / 2;
  const int E_ff = in_sizes[4] / 2;
  const int E_fo = in_sizes[5] / 2;
  const int E_of = in_sizes[6] / 2;
  const int E_fi = in_sizes[7] / 2;

  const float* x_input    = (const float*)d_in[0];
  const float* x_function = (const float*)d_in[1];
  const float* x_output   = (const float*)d_in[2];
  const int* ei_if = (const int*)d_in[3];
  const int* ei_ff = (const int*)d_in[4];
  const int* ei_fo = (const int*)d_in[5];
  const int* ei_of = (const int*)d_in[6];
  const int* ei_fi = (const int*)d_in[7];
  const int* fbatch = (const int*)d_in[8];

  auto P = [&](int i) { return (const float*)d_in[i]; };
  // params: layers (5 x 24 arrays) start at 9; then ln_g, ln_b, att_W, att_b, mu(6), pi(6)
  const int PBASE = 9;
  const float* ln_g  = P(PBASE + 120 + 0);
  const float* ln_b  = P(PBASE + 120 + 1);
  const float* att_W = P(PBASE + 120 + 2);
  const float* att_b = P(PBASE + 120 + 3);
  const float* muW1 = P(PBASE + 124), *mub1 = P(PBASE + 125);
  const float* muW2 = P(PBASE + 126), *mub2 = P(PBASE + 127);
  const float* muW3 = P(PBASE + 128), *mub3 = P(PBASE + 129);
  const float* piW1 = P(PBASE + 130), *pib1 = P(PBASE + 131);
  const float* piW2 = P(PBASE + 132), *pib2 = P(PBASE + 133);
  const float* piW3 = P(PBASE + 134), *pib3 = P(PBASE + 135);

  // ----- workspace layout -----
  char* wsb = (char*)d_ws;
  size_t off = 0;
  auto alloc = [&](size_t nfloats) {
    float* p = (float*)(wsb + off);
    off += ((nfloats * sizeof(float) + 255) / 256) * 256;
    return p;
  };
  float* xi = alloc((size_t)Ni * HC);
  float* xf = alloc((size_t)Nf * HC);
  float* xo = alloc((size_t)No * HC);
  float* comb_i = alloc((size_t)Ni * HC);
  float* comb_f = alloc((size_t)Nf * HC);
  float* comb_o = alloc((size_t)No * HC);
  float* hs = alloc((size_t)Nf * HC);
  float* hd = alloc((size_t)Nf * HC);
  float* ss = alloc(Nf);
  float* sd = alloc(Nf);
  float* mbuf = alloc(Nf);
  float* den = alloc(Nf);
  float* unnorm = alloc((size_t)Nf * HC);
  float* attl = alloc((size_t)Nf * NHEADS);
  float* hmax = alloc(NHEADS);
  float* hsum = alloc(NHEADS);
  float* pooled = alloc(NB * NHEADS * HC);
  float* t1 = alloc(NB * 50);
  float* t2 = alloc(NB * 10);
  float* muv = alloc(NB);
  float* piv = alloc(NB * 2);
  (void)ws_size;

  // ----- init persistent node states -----
  hipMemcpyAsync(xi, x_input,    (size_t)Ni * HC * sizeof(float), hipMemcpyDeviceToDevice, stream);
  hipMemcpyAsync(xf, x_function, (size_t)Nf * HC * sizeof(float), hipMemcpyDeviceToDevice, stream);
  hipMemcpyAsync(xo, x_output,   (size_t)No * HC * sizeof(float), hipMemcpyDeviceToDevice, stream);

  // ----- 5 GAT layers -----
  for (int L = 0; L < 5; ++L) {
    const int o = PBASE + L * 24;
    // per-layer param offsets (dict insertion order: i2f, f2f, f2o, o2f, f2i)
    const float *i2f_Ws = P(o+0),  *i2f_Wd = P(o+1),  *i2f_as = P(o+2),  *i2f_ad = P(o+3),  *i2f_b = P(o+4);
    const float *f2f_W  = P(o+5),  *f2f_as = P(o+6),  *f2f_ad = P(o+7),  *f2f_b = P(o+8);
    const float *f2o_Ws = P(o+9),  *f2o_Wd = P(o+10), *f2o_as = P(o+11), *f2o_ad = P(o+12), *f2o_b = P(o+13);
    const float *o2f_Ws = P(o+14), *o2f_Wd = P(o+15), *o2f_as = P(o+16), *o2f_ad = P(o+17), *o2f_b = P(o+18);
    const float *f2i_Ws = P(o+19), *f2i_Wd = P(o+20), *f2i_as = P(o+21), *f2i_ad = P(o+22), *f2i_b = P(o+23);

    fill_f32<<<((size_t)Ni * HC + 255) / 256, 256, 0, stream>>>(comb_i, 0.0f, Ni * HC);
    fill_f32<<<((size_t)Nf * HC + 255) / 256, 256, 0, stream>>>(comb_f, 0.0f, Nf * HC);
    fill_f32<<<((size_t)No * HC + 255) / 256, 256, 0, stream>>>(comb_o, 0.0f, No * HC);

    run_gat(stream, xi, Ni, xf, Nf, ei_if, E_if, i2f_Ws, i2f_Wd, i2f_as, i2f_ad, i2f_b,
            comb_f, hs, hd, ss, sd, mbuf, den, unnorm);
    run_gat(stream, xf, Nf, xf, Nf, ei_ff, E_ff, f2f_W, f2f_W, f2f_as, f2f_ad, f2f_b,
            comb_f, hs, hd, ss, sd, mbuf, den, unnorm);
    run_gat(stream, xo, No, xf, Nf, ei_of, E_of, o2f_Ws, o2f_Wd, o2f_as, o2f_ad, o2f_b,
            comb_f, hs, hd, ss, sd, mbuf, den, unnorm);
    run_gat(stream, xf, Nf, xo, No, ei_fo, E_fo, f2o_Ws, f2o_Wd, f2o_as, f2o_ad, f2o_b,
            comb_o, hs, hd, ss, sd, mbuf, den, unnorm);
    run_gat(stream, xf, Nf, xi, Ni, ei_fi, E_fi, f2i_Ws, f2i_Wd, f2i_as, f2i_ad, f2i_b,
            comb_i, hs, hd, ss, sd, mbuf, den, unnorm);

    elu_ln<<<((size_t)Nf * 32 + 255) / 256, 256, 0, stream>>>(xf, comb_f, ln_g, ln_b, Nf);
    elu_ln<<<((size_t)No * 32 + 255) / 256, 256, 0, stream>>>(xo, comb_o, ln_g, ln_b, No);
    elu_ln<<<((size_t)Ni * 32 + 255) / 256, 256, 0, stream>>>(xi, comb_i, ln_g, ln_b, Ni);
  }

  // ----- attention pooling (softmax over ALL function nodes per head) -----
  fill_f32<<<1, 32, 0, stream>>>(hmax, -__builtin_inff(), NHEADS);
  fill_f32<<<1, 32, 0, stream>>>(hsum, 0.0f, NHEADS);
  fill_f32<<<(NB * NHEADS * HC + 255) / 256, 256, 0, stream>>>(pooled, 0.0f, NB * NHEADS * HC);
  att_logits<<<(Nf + 255) / 256, 256, 0, stream>>>(xf, att_W, att_b, attl, hmax, Nf);
  att_exp<<<((size_t)Nf * NHEADS + 255) / 256, 256, 0, stream>>>(attl, hmax, hsum, Nf * NHEADS);
  att_pool<<<((size_t)Nf * 32 + 255) / 256, 256, 0, stream>>>(attl, hsum, xf, fbatch, pooled, Nf);

  // ----- MLP heads -----
  const int D = NHEADS * HC;  // 160
  dense<<<(NB * 50 + 255) / 256, 256, 0, stream>>>(pooled, muW1, mub1, t1, NB, D, 50, 1);
  dense<<<(NB * 10 + 255) / 256, 256, 0, stream>>>(t1, muW2, mub2, t2, NB, 50, 10, 1);
  dense<<<(NB * 1 + 255) / 256, 256, 0, stream>>>(t2, muW3, mub3, muv, NB, 10, 1, 0);
  dense<<<(NB * 50 + 255) / 256, 256, 0, stream>>>(pooled, piW1, pib1, t1, NB, D, 50, 1);
  dense<<<(NB * 10 + 255) / 256, 256, 0, stream>>>(t1, piW2, pib2, t2, NB, 50, 10, 1);
  dense<<<(NB * 2 + 255) / 256, 256, 0, stream>>>(t2, piW3, pib3, piv, NB, 10, 2, 0);

  final_out<<<1, NB, 0, stream>>>(muv, piv, (float*)d_out);
  (void)out_size; (void)n_in;
}
